// MRM_17222818857208
// MI455X (gfx1250) — matrix operand order repaired and checked by reading, not compiled, not run on hardware
//
#include <hip/hip_runtime.h>
#include <hip/hip_bf16.h>

typedef __attribute__((ext_vector_type(16))) __bf16 v16bf;
typedef __attribute__((ext_vector_type(8)))  float  v8f;
typedef __attribute__((ext_vector_type(4)))  int    v4i;

#define NB   8
#define NC   256
#define NH   64
#define NW   64
#define NO   256
#define NPIX 4096

#if defined(__has_builtin)
#  if __has_builtin(__builtin_amdgcn_global_load_async_to_lds_b128)
#    define USE_ASYNC_LDS 1
#  endif
#endif
#ifndef USE_ASYNC_LDS
#  define USE_ASYNC_LDS 0
#endif

#define AS1 __attribute__((address_space(1)))
#define AS3 __attribute__((address_space(3)))

#if USE_ASYNC_LDS
__device__ __forceinline__ void async_b128(const void* g, void* l) {
    __builtin_amdgcn_global_load_async_to_lds_b128(
        (AS1 v4i*)(g), (AS3 v4i*)(l), 0, 0);
}
__device__ __forceinline__ void wait_async0() {
#if __has_builtin(__builtin_amdgcn_s_wait_asynccnt)
    __builtin_amdgcn_s_wait_asynccnt(0);
#else
    asm volatile("s_wait_asynccnt 0" ::: "memory");
#endif
}
#endif

union FragU {
    uint4 u[2];
    v16bf v;
};

// ---------------------------------------------------------------------------
// Kernel 1: gates[b][e] = softmax_e( sum_c mean_hw(y[b,c]) * (gw[c,e]+gw[c+256,e]) + gb[e] )
// ---------------------------------------------------------------------------
__global__ __launch_bounds__(256) void gates_kernel(
    const float* __restrict__ y, const float* __restrict__ gw,
    const float* __restrict__ gb, float* __restrict__ gates)
{
    __shared__ float ctx[NC];
    __shared__ float red[3][NC];
    const int b = blockIdx.x;
    const int t = threadIdx.x;

    const float4* p = (const float4*)(y + ((size_t)b * NC + t) * NPIX);
    float s = 0.f;
    for (int i = 0; i < NPIX / 4; ++i) {
        float4 v = p[i];
        s += v.x + v.y + v.z + v.w;
    }
    ctx[t] = s * (1.0f / (float)NPIX);
    __syncthreads();

    const float c = ctx[t];
#pragma unroll
    for (int e = 0; e < 3; ++e)
        red[e][t] = c * (gw[t * 3 + e] + gw[(t + 256) * 3 + e]);
    __syncthreads();

    for (int stride = 128; stride > 0; stride >>= 1) {
        if (t < stride) {
#pragma unroll
            for (int e = 0; e < 3; ++e) red[e][t] += red[e][t + stride];
        }
        __syncthreads();
    }
    if (t == 0) {
        float l0 = red[0][0] + gb[0];
        float l1 = red[1][0] + gb[1];
        float l2 = red[2][0] + gb[2];
        float m  = fmaxf(l0, fmaxf(l1, l2));
        float e0 = __expf(l0 - m), e1 = __expf(l1 - m), e2 = __expf(l2 - m);
        float inv = 1.0f / (e0 + e1 + e2);
        gates[b * 3 + 0] = e0 * inv;
        gates[b * 3 + 1] = e1 * inv;
        gates[b * 3 + 2] = e2 * inv;
    }
}

// ---------------------------------------------------------------------------
// Kernel 2: wmix[b][g][o][c] = sum_e gates[b,e]*(experts[e,o,c,g] + experts[e,o,c+256,g])
// experts layout: [3][256][512][3][3]; g = kh*3+kw
// ---------------------------------------------------------------------------
__global__ __launch_bounds__(256) void mix_kernel(
    const float* __restrict__ experts, const float* __restrict__ gates,
    __hip_bfloat16* __restrict__ wmix)
{
    const int bo = blockIdx.x;          // b*256 + o
    const int b  = bo >> 8;
    const int o  = bo & 255;
    const int c  = threadIdx.x;

    const float g0 = gates[b * 3 + 0];
    const float g1 = gates[b * 3 + 1];
    const float g2 = gates[b * 3 + 2];

#pragma unroll
    for (int g = 0; g < 9; ++g) {
        float acc = 0.f;
#pragma unroll
        for (int e = 0; e < 3; ++e) {
            const size_t base = ((size_t)e * 256 + o) * 512;
            const float v = experts[(base + c) * 9 + g] +
                            experts[(base + c + 256) * 9 + g];
            acc += (e == 0 ? g0 : (e == 1 ? g1 : g2)) * v;
        }
        wmix[(((size_t)(b * 9 + g)) * NO + o) * NC + c] = __float2bfloat16(acc);
    }
}

// ---------------------------------------------------------------------------
// Kernel 3: q [B][C][H][W] fp32  ->  qn [B][H][W][C] bf16
// ---------------------------------------------------------------------------
__global__ __launch_bounds__(256) void nhwc_kernel(
    const float* __restrict__ q, __hip_bfloat16* __restrict__ qn)
{
    const int bh = blockIdx.x;          // b*64 + h
    const int b  = bh >> 6;
    const int h  = bh & 63;
    const int c  = threadIdx.x;

    const float* src = q + (((size_t)b * NC + c) * NH + h) * NW;
    __hip_bfloat16* dst = qn + ((size_t)b * NPIX + h * NW) * NC + c;
    for (int w = 0; w < NW; ++w)
        dst[w * NC] = __float2bfloat16(src[w]);
}

// ---------------------------------------------------------------------------
// Kernel 4: implicit-GEMM dynamic conv via bf16 WMMA, double-buffered LDS with
// CDNA5 async global->LDS tile staging (ASYNCcnt) when available.
// Per batch b: out[o][pix] = sum_g sum_c wmix[b][g][o][c] * qn[b][pix shifted by g][c]
// Block: 128(M=O) x 128(N=pix) tile, 8 waves in 2x4, each wave 64x32 (4x2 WMMA tiles).
// 72 pipeline steps = 9 taps x 8 k-slices of 32; 1 barrier per step.
// ---------------------------------------------------------------------------
#define LDS_ROW 80                 // 32 bf16 = 64B data, padded to 80B (16B aligned)
#define TILE_BYTES (128 * LDS_ROW) // 10240 B per (A or B) tile
#define BUF_BYTES  (2 * TILE_BYTES)
#define NSTEP 72

struct TileAddr {
    const char* arow;   // global src for A 32B segment
    const char* qrow;   // global src for B 32B segment
    bool valid;
};

__device__ __forceinline__ TileAddr tile_addr(
    const __hip_bfloat16* __restrict__ wmix, const __hip_bfloat16* __restrict__ qn,
    int b, int mBase, int nBase, int r, int seg, int it)
{
    const int g   = it >> 3;
    const int ks  = it & 7;
    const int c0b = ks * 64;                 // byte offset of 32-channel k-slice
    const int dy  = g / 3 - 1;
    const int dx  = g % 3 - 1;
    const int pix = nBase + r;
    const int ph  = (pix >> 6) + dy;
    const int pw  = (pix & 63) + dx;
    TileAddr t;
    t.valid = (ph >= 0) & (ph < NH) & (pw >= 0) & (pw < NW);
    t.arow  = (const char*)(wmix +
        (((size_t)(b * 9 + g)) * NO + (mBase + r)) * NC) + c0b + seg;
    t.qrow  = (const char*)(qn +
        ((size_t)b * NPIX + ph * NW + pw) * NC) + c0b + seg;
    return t;
}

__global__ __launch_bounds__(256) void dynconv_wmma(
    const __hip_bfloat16* __restrict__ wmix,   // [B][9][O][C]
    const __hip_bfloat16* __restrict__ qn,     // [B][H][W][C]
    float* __restrict__ out)                   // [B][O][H*W]
{
    // two double-buffers: [buf][A tile | B tile]
    __shared__ uint4 smem4[(2 * BUF_BYTES) / 16];   // 40960 B
    char* smem = (char*)smem4;

    const int b     = blockIdx.z;
    const int mBase = blockIdx.y * 128;     // O base
    const int nBase = blockIdx.x * 128;     // pixel base
    const int tid   = threadIdx.x;
    const int lane  = tid & 31;
    const int wave  = tid >> 5;
    const int wm    = wave >> 2;            // 0..1  (64 rows of M each)
    const int wn    = wave & 3;             // 0..3  (32 cols of N each)
    const int lrow  = lane & 15;
    const int hiH   = lane >> 4;            // 0/1
    // A fragment byte offsets within 64B row (documented 16-bit A layout)
    const int aOff0 = hiH * 16;             // K 0..7  / 8..15
    const int aOff1 = 32 + hiH * 16;        // K 16..23 / 24..31
    // B fragment byte offsets (documented 16-bit B layout)
    const int bOff0 = hiH * 16;             // K 0..7  / 8..15
    const int bOff1 = 32 + hiH * 16;        // K 16..23 / 24..31

    v8f acc[4][2];
#pragma unroll
    for (int m = 0; m < 4; ++m)
#pragma unroll
        for (int n = 0; n < 2; ++n)
            acc[m][n] = (v8f){0.f, 0.f, 0.f, 0.f, 0.f, 0.f, 0.f, 0.f};

    // staging assignment: each thread owns 32B of one 64B row
    const int r   = tid >> 1;               // 0..127
    const int seg = (tid & 1) * 32;         // 0 or 32 bytes
    const int stA = r * LDS_ROW + seg;      // store offset within A tile
    const int stB = TILE_BYTES + r * LDS_ROW + seg;

#if USE_ASYNC_LDS
    // ---- stage a tile with async global->LDS DMA; zero-fill padding via DS ----
    auto stage = [&](int it, char* buf) {
        TileAddr t = tile_addr(wmix, qn, b, mBase, nBase, r, seg, it);
        async_b128(t.arow,      buf + stA);
        async_b128(t.arow + 16, buf + stA + 16);
        if (t.valid) {
            async_b128(t.qrow,      buf + stB);
            async_b128(t.qrow + 16, buf + stB + 16);
        } else {
            *(uint4*)(buf + stB)      = (uint4){0u, 0u, 0u, 0u};
            *(uint4*)(buf + stB + 16) = (uint4){0u, 0u, 0u, 0u};
        }
    };

    stage(0, smem);            // prologue -> buffer 0
    wait_async0();
    __syncthreads();

#pragma unroll 2
    for (int it = 0; it < NSTEP; ++it) {
        char* curA = smem + (it & 1) * BUF_BYTES;
        char* curB = curA + TILE_BYTES;

        // issue next tile's DMA into the other buffer (its readers passed the
        // barrier at the end of iteration it-1)
        if (it + 1 < NSTEP)
            stage(it + 1, smem + ((it + 1) & 1) * BUF_BYTES);

        // ---- fragments + WMMA on current buffer ----
        FragU af[4], bf[2];
#pragma unroll
        for (int m = 0; m < 4; ++m) {
            const int ar = wm * 64 + m * 16 + lrow;
            af[m].u[0] = *(const uint4*)(curA + ar * LDS_ROW + aOff0);
            af[m].u[1] = *(const uint4*)(curA + ar * LDS_ROW + aOff1);
        }
#pragma unroll
        for (int n = 0; n < 2; ++n) {
            const int br = wn * 32 + n * 16 + lrow;
            bf[n].u[0] = *(const uint4*)(curB + br * LDS_ROW + bOff0);
            bf[n].u[1] = *(const uint4*)(curB + br * LDS_ROW + bOff1);
        }
#pragma unroll
        for (int m = 0; m < 4; ++m)
#pragma unroll
            for (int n = 0; n < 2; ++n)
                acc[m][n] = __builtin_amdgcn_wmma_f32_16x16x32_bf16(
                    false, af[m].v, false, bf[n].v,
                    (short)0, acc[m][n], false, false);

        wait_async0();          // next tile landed in LDS
        __syncthreads();
    }
#else
    // ---- fallback: register-staged double buffering ----
    {
        TileAddr t0 = tile_addr(wmix, qn, b, mBase, nBase, r, seg, 0);
        uint4 a0 = ((const uint4*)t0.arow)[0];
        uint4 a1 = ((const uint4*)t0.arow)[1];
        uint4 b0 = {0u,0u,0u,0u}, b1 = {0u,0u,0u,0u};
        if (t0.valid) { b0 = ((const uint4*)t0.qrow)[0]; b1 = ((const uint4*)t0.qrow)[1]; }
        *(uint4*)(smem + stA)      = a0;
        *(uint4*)(smem + stA + 16) = a1;
        *(uint4*)(smem + stB)      = b0;
        *(uint4*)(smem + stB + 16) = b1;
    }
    __syncthreads();

#pragma unroll 2
    for (int it = 0; it < NSTEP; ++it) {
        char* curA = smem + (it & 1) * BUF_BYTES;
        char* curB = curA + TILE_BYTES;
        char* nxt  = smem + ((it + 1) & 1) * BUF_BYTES;

        uint4 a0, a1, b0 = {0u,0u,0u,0u}, b1 = {0u,0u,0u,0u};
        const bool more = (it + 1) < NSTEP;
        if (more) {
            TileAddr t = tile_addr(wmix, qn, b, mBase, nBase, r, seg, it + 1);
            a0 = ((const uint4*)t.arow)[0];
            a1 = ((const uint4*)t.arow)[1];
            if (t.valid) { b0 = ((const uint4*)t.qrow)[0]; b1 = ((const uint4*)t.qrow)[1]; }
        }

        FragU af[4], bf[2];
#pragma unroll
        for (int m = 0; m < 4; ++m) {
            const int ar = wm * 64 + m * 16 + lrow;
            af[m].u[0] = *(const uint4*)(curA + ar * LDS_ROW + aOff0);
            af[m].u[1] = *(const uint4*)(curA + ar * LDS_ROW + aOff1);
        }
#pragma unroll
        for (int n = 0; n < 2; ++n) {
            const int br = wn * 32 + n * 16 + lrow;
            bf[n].u[0] = *(const uint4*)(curB + br * LDS_ROW + bOff0);
            bf[n].u[1] = *(const uint4*)(curB + br * LDS_ROW + bOff1);
        }
#pragma unroll
        for (int m = 0; m < 4; ++m)
#pragma unroll
            for (int n = 0; n < 2; ++n)
                acc[m][n] = __builtin_amdgcn_wmma_f32_16x16x32_bf16(
                    false, af[m].v, false, bf[n].v,
                    (short)0, acc[m][n], false, false);

        if (more) {
            *(uint4*)(nxt + stA)      = a0;
            *(uint4*)(nxt + stA + 16) = a1;
            *(uint4*)(nxt + stB)      = b0;
            *(uint4*)(nxt + stB + 16) = b1;
        }
        __syncthreads();
    }
#endif

    // ---- store: C/D layout M = vgpr + 8*hiH, N = lane&15 ----
#pragma unroll
    for (int m = 0; m < 4; ++m) {
#pragma unroll
        for (int n = 0; n < 2; ++n) {
            const int pxl = nBase + wn * 32 + n * 16 + lrow;
#pragma unroll
            for (int v = 0; v < 8; ++v) {
                const int o = mBase + wm * 64 + m * 16 + v + hiH * 8;
                out[((size_t)b * NO + o) * NPIX + pxl] = acc[m][n][v];
            }
        }
    }
}

// ---------------------------------------------------------------------------
extern "C" void kernel_launch(void* const* d_in, const int* in_sizes, int n_in,
                              void* d_out, int out_size, void* d_ws, size_t ws_size,
                              hipStream_t stream)
{
    const float* q       = (const float*)d_in[0];
    const float* y       = (const float*)d_in[1];
    const float* experts = (const float*)d_in[2];
    const float* gate_w  = (const float*)d_in[3];
    const float* gate_b  = (const float*)d_in[4];
    float* out = (float*)d_out;

    char* ws = (char*)d_ws;
    float* gates = (float*)ws;                                       // 96 B
    __hip_bfloat16* wmix = (__hip_bfloat16*)(ws + 256);              // 8*9*256*256*2 = 9.44 MB
    __hip_bfloat16* qn   = (__hip_bfloat16*)(ws + 256 +
                               (size_t)NB * 9 * NO * NC * 2);        // 8*4096*256*2 = 16.8 MB

    gates_kernel<<<NB, 256, 0, stream>>>(y, gate_w, gate_b, gates);
    mix_kernel<<<NB * NO, 256, 0, stream>>>(experts, gates, wmix);
    nhwc_kernel<<<NB * NH, 256, 0, stream>>>(q, qn);

    dim3 grid(NPIX / 128, NO / 128, NB);   // 32 x 2 x 8
    dynconv_wmma<<<grid, 256, 0, stream>>>(wmix, qn, out);
}
